// LSTMEncoder_25709674234378
// MI455X (gfx1250) — matrix operand order repaired and checked by reading, not compiled, not run on hardware
//
#include <hip/hip_runtime.h>

// ---------------------------------------------------------------------------
// LSTM encoder for MI455X (gfx1250, wave32)
//   T=8192, I=1024, H=1024, 4H=4096
// Phase 1: x_proj = X @ W_ih^T + (b_ih+b_hh)   -- bf16 WMMA GEMM
// Phase 2: persistent-kernel recurrence, register-resident W_hh slices
//          (64 dwords/thread), wave32 shuffle reduction, device-wide
//          barrier per timestep.
// ---------------------------------------------------------------------------

typedef __attribute__((ext_vector_type(16))) __bf16 v16bf;
typedef __attribute__((ext_vector_type(8)))  float  v8f;

typedef unsigned short u16;
typedef unsigned int   u32;

#define T_STEPS 8192
#define I_DIM   1024
#define H_DIM   1024
#define G_DIM   4096      // 4*H
#define NWG     64        // persistent workgroups in recurrence
#define HC      16        // hidden units owned per workgroup (H/NWG)
#define RB      512       // threads per recurrence block (16 waves)

// ---------------- workspace layout (bytes) ----------------
#define OFF_ABF   ((size_t)0)           // bf16 input      16 MiB
#define OFF_WIH   ((size_t)16777216)    // bf16 W_ih        8 MiB
#define OFF_WHH   ((size_t)25165824)    // bf16 W_hh        8 MiB
#define OFF_XPROJ ((size_t)33554432)    // f32 [T][4H]    128 MiB
#define OFF_HBUF  ((size_t)167772160)   // f32 [2][H] ping-pong
#define OFF_SYNC  (OFF_HBUF + 8192)     // u32 barrier counter

// ---------------- fp32 -> bf16 (round to nearest even) ----------------
__global__ void cvt_f32_bf16(const float* __restrict__ src,
                             u16* __restrict__ dst, int n) {
  int i = blockIdx.x * blockDim.x + threadIdx.x;
  if (i < n) {
    u32 u = __float_as_uint(src[i]);
    u32 r = 0x7fffu + ((u >> 16) & 1u);
    dst[i] = (u16)((u + r) >> 16);
  }
}

// ---------------- zero h0 and barrier counter ----------------
__global__ void init_state(float* __restrict__ hbuf, u32* __restrict__ sync) {
  int i = blockIdx.x * blockDim.x + threadIdx.x;
  if (i < 2 * H_DIM) hbuf[i] = 0.0f;
  if (i == 0) *sync = 0u;
}

// ---------------- Phase 1: WMMA GEMM  xproj = X @ W_ih^T + bias ----------
// Block = 256 threads (8 waves: 2 M-waves x 4 N-waves) -> block tile 32x256.
// Each wave computes a 16x64 tile (A fragment reused across 4 B fragments).
union FragBf { uint4 q[2]; v16bf v; };

__global__ __launch_bounds__(256, 1)
void gemm_xproj(const u16* __restrict__ Abf, const u16* __restrict__ Wbf,
                const float* __restrict__ b_ih, const float* __restrict__ b_hh,
                float* __restrict__ xproj) {
  const int tid  = threadIdx.x;
  const int lane = tid & 31;
  const int wave = tid >> 5;
  const int wm = wave & 1;          // 0..1
  const int wn = wave >> 1;         // 0..3
  const int M0 = blockIdx.x * 32 + wm * 16;
  const int N0 = blockIdx.y * 256 + wn * 64;
  const int r  = lane & 15;
  const int hi = lane >> 4;         // half-wave selector

  // A-fragment rows: lane holds row M0+r; K chunks at (hi*8) and (16+hi*8)
  const uint4* Arow = (const uint4*)(const void*)(Abf + (size_t)(M0 + r) * I_DIM);

  v8f acc[4];
  #pragma unroll
  for (int n = 0; n < 4; ++n) acc[n] = (v8f){};

  for (int k0 = 0; k0 < I_DIM; k0 += 32) {
    FragBf a;
    a.q[0] = Arow[(k0 >> 3) + hi];       // K = k0 + hi*8      .. +7
    a.q[1] = Arow[(k0 >> 3) + hi + 2];   // K = k0 + 16 + hi*8 .. +7
    #pragma unroll
    for (int n = 0; n < 4; ++n) {
      // B-fragment: lane holds column N0+n*16+r; K chunks at (hi*8) and (16+hi*8)
      const uint4* Wrow =
          (const uint4*)(const void*)(Wbf + (size_t)(N0 + n * 16 + r) * I_DIM);
      FragBf b;
      b.q[0] = Wrow[(k0 >> 3) + hi];
      b.q[1] = Wrow[(k0 >> 3) + hi + 2];
      acc[n] = __builtin_amdgcn_wmma_f32_16x16x32_bf16(
          false, a.v, false, b.v, (short)0, acc[n], false, false);
    }
  }

  #pragma unroll
  for (int n = 0; n < 4; ++n) {
    const int col = N0 + n * 16 + r;
    const float bias = b_ih[col] + b_hh[col];
    #pragma unroll
    for (int e = 0; e < 8; ++e) {
      const int m = M0 + hi * 8 + e;   // C/D layout: VGPR e -> M=e (+8 upper half)
      xproj[(size_t)m * G_DIM + col] = acc[n][e] + bias;
    }
  }
}

// ---------------- Phase 2: persistent recurrence ----------------
__device__ __forceinline__ float sigmoidf_(float x) {
  return 1.0f / (1.0f + __expf(-x));
}
__device__ __forceinline__ float tanhf_(float x) {
  x = fminf(fmaxf(x, -15.0f), 15.0f);
  float e = __expf(-2.0f * x);
  return (1.0f - e) / (1.0f + e);
}

__global__ __launch_bounds__(RB, 1)
void lstm_recurrent(const u16* __restrict__ Whh_bf,
                    const float* __restrict__ xproj,
                    float* __restrict__ hbuf,          // [2][1024] ping-pong
                    u32* __restrict__ sync,
                    float* __restrict__ out) {         // [T][H]
  __shared__ float hl[H_DIM];       // broadcast h_t
  __shared__ float part[64];        // one full dot product per gate row
  __shared__ float gatesl[64];      // i,f,g,o for our 16 hidden units
  __shared__ float cl[HC];          // cell state slice

  const int tid = threadIdx.x;      // 0..511
  const int g   = blockIdx.x;       // 0..NWG-1
  const int r   = tid >> 3;         // slice gate-row 0..63
  const int p   = tid & 7;          // K-eighth 0..7

  // Register-resident W_hh slice: this thread's 128 bf16 weights (64 dwords).
  // slice row r -> global gate row R = (r/16)*1024 + g*16 + (r%16)
  const int R = (r >> 4) * H_DIM + g * HC + (r & 15);
  const u32* __restrict__ Wsrc32 =
      (const u32*)(const void*)Whh_bf + (size_t)R * (I_DIM / 2) + p * 64;
  u32 wreg[64];
  #pragma unroll
  for (int kk = 0; kk < 64; ++kk) wreg[kk] = Wsrc32[kk];

  if (tid < HC) cl[tid] = 0.0f;
  __syncthreads();

  for (int t = 0; t < T_STEPS; ++t) {
    const int cur = t & 1;

    // broadcast h_{t} (written last step / init) into LDS
    const float* __restrict__ hs = hbuf + cur * H_DIM;
    for (int i = tid; i < H_DIM; i += RB) hl[i] = hs[i];
    __syncthreads();

    // GEMV: 8 threads per gate row, 128 K each (weights in VGPRs)
    float acc = 0.0f;
    const int kbase = p * 128;
    #pragma unroll
    for (int kk = 0; kk < 64; ++kk) {
      const u32 u = wreg[kk];
      const float w0 = __uint_as_float(u << 16);
      const float w1 = __uint_as_float(u & 0xffff0000u);
      acc = fmaf(w0, hl[kbase + 2 * kk],     acc);
      acc = fmaf(w1, hl[kbase + 2 * kk + 1], acc);
    }
    // wave32 shuffle reduction across the 8 adjacent K-part lanes
    acc += __shfl_xor(acc, 1);
    acc += __shfl_xor(acc, 2);
    acc += __shfl_xor(acc, 4);
    if (p == 0) part[r] = acc;
    __syncthreads();

    if (tid < 64) {
      const int Rt = (tid >> 4) * H_DIM + g * HC + (tid & 15);
      gatesl[tid] = part[tid] + xproj[(size_t)t * G_DIM + Rt];
      if (t + 1 < T_STEPS)   // CDNA5 global_prefetch for next step's x_proj
        __builtin_prefetch(&xproj[(size_t)(t + 1) * G_DIM + Rt], 0, 3);
    }
    __syncthreads();

    if (tid < HC) {
      const float iv = sigmoidf_(gatesl[tid]);
      const float fv = sigmoidf_(gatesl[16 + tid]);
      const float gv = tanhf_   (gatesl[32 + tid]);
      const float ov = sigmoidf_(gatesl[48 + tid]);
      const float c  = fv * cl[tid] + iv * gv;
      cl[tid] = c;
      const float h = ov * tanhf_(c);
      const int hidx = g * HC + tid;
      hbuf[(cur ^ 1) * H_DIM + hidx] = h;   // next step's input buffer
      out[(size_t)t * H_DIM + hidx]  = h;
      __threadfence();                       // make h visible device-wide
    }
    __syncthreads();

    // device-wide barrier: monotonic counter, target = (t+1)*NWG
    if (tid == 0) {
      __hip_atomic_fetch_add(sync, 1u, __ATOMIC_RELEASE,
                             __HIP_MEMORY_SCOPE_AGENT);
      const u32 target = (u32)(t + 1) * NWG;
      while (__hip_atomic_load(sync, __ATOMIC_ACQUIRE,
                               __HIP_MEMORY_SCOPE_AGENT) < target)
        __builtin_amdgcn_s_sleep(2);
    }
    __syncthreads();
  }
}

// ---------------------------------------------------------------------------
extern "C" void kernel_launch(void* const* d_in, const int* in_sizes, int n_in,
                              void* d_out, int out_size, void* d_ws, size_t ws_size,
                              hipStream_t stream) {
  const float* input = (const float*)d_in[0];   // [8192,1024]
  const float* W_ih  = (const float*)d_in[1];   // [4096,1024]
  const float* W_hh  = (const float*)d_in[2];   // [4096,1024]
  const float* b_ih  = (const float*)d_in[3];   // [4096]
  const float* b_hh  = (const float*)d_in[4];   // [4096]
  float* out = (float*)d_out;                   // [8192,1024]

  char* ws = (char*)d_ws;
  u16*   Abf    = (u16*)(ws + OFF_ABF);
  u16*   Wih_bf = (u16*)(ws + OFF_WIH);
  u16*   Whh_bf = (u16*)(ws + OFF_WHH);
  float* xproj  = (float*)(ws + OFF_XPROJ);
  float* hbuf   = (float*)(ws + OFF_HBUF);
  u32*   sync   = (u32*)(ws + OFF_SYNC);

  // 1) fp32 -> bf16 copies for WMMA / register-resident weights
  {
    int n = T_STEPS * I_DIM;
    cvt_f32_bf16<<<(n + 255) / 256, 256, 0, stream>>>(input, Abf, n);
  }
  {
    int n = G_DIM * I_DIM;
    cvt_f32_bf16<<<(n + 255) / 256, 256, 0, stream>>>(W_ih, Wih_bf, n);
    cvt_f32_bf16<<<(n + 255) / 256, 256, 0, stream>>>(W_hh, Whh_bf, n);
  }

  // 2) big input-projection GEMM with bf16 WMMA
  {
    dim3 grid(T_STEPS / 32, G_DIM / 256);
    gemm_xproj<<<grid, 256, 0, stream>>>(Abf, Wih_bf, b_ih, b_hh, xproj);
  }

  // 3) reset h0/c0 and barrier counter (deterministic per launch)
  init_state<<<8, 256, 0, stream>>>(hbuf, sync);

  // 4) persistent sequential recurrence
  lstm_recurrent<<<NWG, RB, 0, stream>>>(Whh_bf, xproj, hbuf, sync, out);
}
